// Block_46471546143558
// MI455X (gfx1250) — compile-verified, matrix operand order checked by reading, not run on hardware
//
#include <hip/hip_runtime.h>
#include <math.h>

typedef unsigned short u16;
typedef __attribute__((ext_vector_type(16))) __bf16 v16bf;
typedef __attribute__((ext_vector_type(8)))  float  v8f;

constexpr int B_ = 2, T_ = 2048, D_ = 1024, H_ = 16, HD_ = 64, E_ = 8, CAP_ = 640;
constexpr int NT = B_ * T_;      // 4096 tokens
constexpr int FF = 4 * D_;       // 4096

#define USE_ASYNC_LDS 1

__device__ __forceinline__ u16 f2bf(float f) {
    unsigned int u = __float_as_uint(f);
    unsigned int r = u + 0x7FFFu + ((u >> 16) & 1u);
    return (u16)(r >> 16);
}

union ABFrag { v16bf v; uint4 q[2]; };

__device__ __forceinline__ v8f wmma_bf16(v16bf a, v16bf b, v8f c) {
    return __builtin_amdgcn_wmma_f32_16x16x32_bf16(false, a, false, b, (short)0, c, false, false);
}

// 16-byte async copy global -> LDS (ASYNCcnt-tracked). LDS address = low 32 bits
// of the generic pointer.
__device__ __forceinline__ void cp16_async(u16* l, const u16* g) {
#if USE_ASYNC_LDS
    unsigned la = (unsigned)(unsigned long long)l;
    asm volatile("global_load_async_to_lds_b128 %0, %1, off" :: "v"(la), "v"(g) : "memory");
#else
    *(uint4*)l = *(const uint4*)g;
#endif
}
__device__ __forceinline__ void wait_async4() {
#if USE_ASYNC_LDS
    asm volatile("s_wait_asynccnt 0x4" ::: "memory");
#endif
}
__device__ __forceinline__ void wait_async0() {
#if USE_ASYNC_LDS
    asm volatile("s_wait_asynccnt 0x0" ::: "memory");
#endif
}

// ---------------- elementwise weight prep ----------------
__global__ void cast_bf_kernel(const float* __restrict__ src, u16* __restrict__ dst, long n) {
    for (long i = (long)blockIdx.x * blockDim.x + threadIdx.x; i < n;
         i += (long)gridDim.x * blockDim.x)
        dst[i] = f2bf(src[i]);
}

// src shape (E, R, C) -> dst shape (E, C, R)
__global__ void transpose_cast_kernel(const float* __restrict__ src, u16* __restrict__ dst,
                                      int Eb, int R, int C) {
    long total = (long)Eb * R * C;
    for (long i = (long)blockIdx.x * blockDim.x + threadIdx.x; i < total;
         i += (long)gridDim.x * blockDim.x) {
        int rr = (int)(i % R);
        int cc = (int)((i / R) % C);
        int e  = (int)(i / ((long)R * C));
        dst[i] = f2bf(src[((long)e * R + rr) * C + cc]);
    }
}

// ---------------- layernorm ----------------
__global__ __launch_bounds__(256) void ln_kernel(const float* __restrict__ x,
                                                 const float* __restrict__ g,
                                                 const float* __restrict__ bta,
                                                 u16* out_bf, float* out_f, int D) {
    int row = blockIdx.x;
    const float* xr = x + (size_t)row * D;
    __shared__ float ws1[8], ws2[8], sm[2];
    int tid = threadIdx.x, lane = tid & 31, wid = tid >> 5;
    float s = 0.f, s2 = 0.f;
    for (int d = tid; d < D; d += 256) { float v = xr[d]; s += v; s2 += v * v; }
    for (int o = 16; o; o >>= 1) { s += __shfl_xor(s, o); s2 += __shfl_xor(s2, o); }
    if (lane == 0) { ws1[wid] = s; ws2[wid] = s2; }
    __syncthreads();
    if (tid == 0) {
        float a = 0.f, b2 = 0.f;
        for (int i = 0; i < 8; i++) { a += ws1[i]; b2 += ws2[i]; }
        float mean = a / D;
        float var  = b2 / D - mean * mean;
        sm[0] = mean; sm[1] = rsqrtf(var + 1e-5f);
    }
    __syncthreads();
    float mean = sm[0], inv = sm[1];
    for (int d = tid; d < D; d += 256) {
        float v = (xr[d] - mean) * inv * g[d] + bta[d];
        if (out_bf) out_bf[(size_t)row * D + d] = f2bf(v);
        if (out_f)  out_f[(size_t)row * D + d]  = v;
    }
}

// ---------------- bf16 NT GEMM with async-LDS double buffering ----------------
// C = A(MxK) * Bw(NxK)^T.  Block = 128 threads (4 waves), 64x64 C tile, K step 32.
// EPI 0: fp32 out (+ optional residual). EPI 1: +bias, exact GELU, bf16 out. EPI 2: +bias, fp32 out.
constexpr int LROW = 40;            // padded LDS row (elements): 80B, 16B-aligned, bank-spread
constexpr int LBUF = 64 * LROW;     // one tile buffer

template <int EPI>
__global__ __launch_bounds__(128) void gemm_nt(const u16* __restrict__ A,
                                               const u16* __restrict__ Bw,
                                               float* Cf, u16* Cb,
                                               const float* __restrict__ bias,
                                               const float* __restrict__ resid,
                                               int N, int Kd,
                                               long sA, long sB, long sC, long sBias) {
    __shared__ __attribute__((aligned(16))) u16 lAs[2][LBUF];
    __shared__ __attribute__((aligned(16))) u16 lBs[2][LBUF];

    int z = blockIdx.z;
    A  += (size_t)z * sA;
    Bw += (size_t)z * sB;
    if (bias) bias += (size_t)z * sBias;
    size_t cOff = (size_t)z * sC;

    int tid = threadIdx.x, lane = tid & 31, wave = tid >> 5;
    int wm = wave >> 1, wn = wave & 1;
    int rowB = blockIdx.y * 64;
    int colB = blockIdx.x * 64;
    int half = lane >> 4, lm = lane & 15, c = half * 8;

    // copy mapping: thread t moves 32B of A and 32B of B per stage
    int rr = tid >> 1;                 // tile row 0..63
    int ce = (tid & 1) * 16;           // element column 0 or 16 within the 32-wide K slab
    const u16* gA = A  + (size_t)(rowB + rr) * Kd + ce;
    const u16* gB = Bw + (size_t)(colB + rr) * Kd + ce;
    u16* lA = &lAs[0][0] + rr * LROW + ce;
    u16* lB = &lBs[0][0] + rr * LROW + ce;

    v8f acc[2][2] = {};
    int nst = Kd / 32;

    // stage 0 issue
    cp16_async(lA, gA);      cp16_async(lA + 8, gA + 8);
    cp16_async(lB, gB);      cp16_async(lB + 8, gB + 8);

    for (int s = 0; s < nst; s++) {
        int cur = s & 1;
        __syncthreads();                       // prev reads of the other buffer done
        if (s + 1 < nst) {
            int nxt = (s + 1) & 1;
            int k = (s + 1) * 32;
            cp16_async(lA + nxt * LBUF, gA + k);
            cp16_async(lA + nxt * LBUF + 8, gA + k + 8);
            cp16_async(lB + nxt * LBUF, gB + k);
            cp16_async(lB + nxt * LBUF + 8, gB + k + 8);
            wait_async4();                     // in-order: current stage's 4 copies done
        } else {
            wait_async0();
        }
        __syncthreads();                       // all waves' copies visible

        const u16* As = &lAs[cur][0];
        const u16* Bs = &lBs[cur][0];
        ABFrag a[2], b[2];
        for (int i = 0; i < 2; i++) {
            const u16* p = As + (wm * 32 + i * 16 + lm) * LROW + c;
            a[i].q[0] = *(const uint4*)p;
            a[i].q[1] = *(const uint4*)(p + 16);
        }
        for (int j = 0; j < 2; j++) {
            const u16* p = Bs + (wn * 32 + j * 16 + lm) * LROW + c;
            b[j].q[0] = *(const uint4*)p;
            b[j].q[1] = *(const uint4*)(p + 16);
        }
        for (int i = 0; i < 2; i++)
            for (int j = 0; j < 2; j++)
                acc[i][j] = wmma_bf16(a[i].v, b[j].v, acc[i][j]);
    }

    for (int i = 0; i < 2; i++)
        for (int j = 0; j < 2; j++)
            for (int r = 0; r < 8; r++) {
                int row = rowB + wm * 32 + i * 16 + r + 8 * half;
                int col = colB + wn * 32 + j * 16 + lm;
                float v = acc[i][j][r];
                size_t idx = cOff + (size_t)row * N + col;
                if (EPI == 0) {
                    if (resid) v += resid[idx];
                    Cf[idx] = v;
                } else if (EPI == 1) {
                    v += bias[col];
                    v = 0.5f * v * (1.0f + erff(v * 0.70710678118654752f));
                    Cb[idx] = f2bf(v);
                } else {
                    v += bias[col];
                    Cf[idx] = v;
                }
            }
}

// ---------------- RoPE + pack q/k/v to bf16 (q pre-scaled by 1/sqrt(HD)) ----------------
__global__ __launch_bounds__(64) void rope_pack(const float* __restrict__ qkv,
                                                u16* qb, u16* kb, u16* vb) {
    int n = blockIdx.x;            // = (b*T + t)*H + h
    int h  = n % H_;
    int bt = n / H_;
    int t  = bt % T_;
    int d  = threadIdx.x;
    const float* base = qkv + (size_t)bt * (3 * D_) + h * (3 * HD_);
    int j = (d < 32) ? d : d - 32;
    float inv = expf(-(float)j * (9.210340371976184f / 32.0f)); // 10000^(-2j/HD)
    float ang = (float)t * inv;
    float sn = sinf(ang), cs = cosf(ang);
    float q1 = base[j],      q2 = base[j + 32];
    float k1 = base[64 + j], k2 = base[96 + j];
    float qr = (d < 32) ? (q1 * cs - q2 * sn) : (q1 * sn + q2 * cs);
    float kr = (d < 32) ? (k1 * cs - k2 * sn) : (k1 * sn + k2 * cs);
    size_t o = ((size_t)bt * H_ + h) * HD_ + d;
    qb[o] = f2bf(qr * 0.125f);
    kb[o] = f2bf(kr);
    vb[o] = f2bf(base[128 + d]);
}

// ---------------- flash attention (causal), WMMA bf16, fp32 softmax ----------------
__global__ __launch_bounds__(128) void flash_attn(const u16* __restrict__ qb,
                                                  const u16* __restrict__ kb,
                                                  const u16* __restrict__ vb,
                                                  u16* __restrict__ ob) {
    __shared__ __attribute__((aligned(16))) u16 vt[64][72];      // V^T tile
    __shared__ __attribute__((aligned(16))) u16 pt[4][16][72];   // per-wave P tile
    int qt = blockIdx.x;
    int bh = blockIdx.y;
    int b = bh >> 4, hh = bh & 15;
    int tid = threadIdx.x, lane = tid & 31, w = tid >> 5;
    int half = lane >> 4, lm = lane & 15, c = half * 8;

    int qrow = qt * 64 + w * 16 + lm;
    const u16* qp = qb + ((size_t)(b * T_ + qrow) * H_ + hh) * HD_ + c;
    ABFrag qa[2];
    for (int s = 0; s < 2; s++) {
        qa[s].q[0] = *(const uint4*)(qp + s * 32);
        qa[s].q[1] = *(const uint4*)(qp + s * 32 + 16);
    }
    float rm[8], rl[8];
    v8f oacc[4] = {};
    for (int r = 0; r < 8; r++) { rm[r] = -3e38f; rl[r] = 0.f; }

    for (int kt = 0; kt <= qt; kt++) {
        int kt0 = kt * 64;
        __syncthreads();
        // stage V tile transposed into LDS
        for (int idx = tid; idx < 4096; idx += 128) {
            int tok = idx >> 6, dim = idx & 63;
            vt[dim][tok] = vb[((size_t)(b * T_ + kt0 + tok) * H_ + hh) * HD_ + dim];
        }
        // S = Q * K^T  (K tile rows in global are directly the NT B-operand)
        v8f sacc[4] = {};
        for (int nf = 0; nf < 4; nf++)
            for (int s = 0; s < 2; s++) {
                ABFrag kf;
                const u16* kp = kb + ((size_t)(b * T_ + kt0 + nf * 16 + lm) * H_ + hh) * HD_ + s * 32 + c;
                kf.q[0] = *(const uint4*)kp;
                kf.q[1] = *(const uint4*)(kp + 16);
                sacc[nf] = wmma_bf16(qa[s].v, kf.v, sacc[nf]);
            }
        if (kt == qt) { // diagonal tile: causal mask
            for (int nf = 0; nf < 4; nf++)
                for (int r = 0; r < 8; r++) {
                    int tq = qt * 64 + w * 16 + r + 8 * half;
                    int tk = kt0 + nf * 16 + lm;
                    if (tk > tq) sacc[nf][r] = -1.0e9f;
                }
        }
        // online softmax
        float mnew[8], scale[8];
        for (int r = 0; r < 8; r++) {
            float m = sacc[0][r];
            for (int nf = 1; nf < 4; nf++) m = fmaxf(m, sacc[nf][r]);
            for (int o = 1; o < 16; o <<= 1) m = fmaxf(m, __shfl_xor(m, o));
            mnew[r]  = fmaxf(rm[r], m);
            scale[r] = expf(rm[r] - mnew[r]);
            rm[r] = mnew[r];
        }
        for (int r = 0; r < 8; r++) {
            float sum = 0.f;
            for (int nf = 0; nf < 4; nf++) {
                float p = expf(sacc[nf][r] - mnew[r]);
                sum += p;
                pt[w][r + 8 * half][nf * 16 + lm] = f2bf(p);
            }
            for (int o = 1; o < 16; o <<= 1) sum += __shfl_xor(sum, o);
            rl[r] = rl[r] * scale[r] + sum;
            for (int nd = 0; nd < 4; nd++) oacc[nd][r] *= scale[r];
        }
        __syncthreads();
        // O += P * V
        for (int s = 0; s < 2; s++) {
            ABFrag pf;
            const u16* pp = &pt[w][lm][0] + s * 32 + c;
            pf.q[0] = *(const uint4*)pp;
            pf.q[1] = *(const uint4*)(pp + 16);
            for (int nd = 0; nd < 4; nd++) {
                ABFrag vf;
                const u16* vp = &vt[nd * 16 + lm][0] + s * 32 + c;
                vf.q[0] = *(const uint4*)vp;
                vf.q[1] = *(const uint4*)(vp + 16);
                oacc[nd] = wmma_bf16(pf.v, vf.v, oacc[nd]);
            }
        }
    }
    for (int nd = 0; nd < 4; nd++)
        for (int r = 0; r < 8; r++) {
            int tq  = qt * 64 + w * 16 + r + 8 * half;
            int dim = hh * HD_ + nd * 16 + lm;
            ob[(size_t)(b * T_ + tq) * D_ + dim] = f2bf(oacc[nd][r] / rl[r]);
        }
}

// ---------------- MoE gating: softmax over E=8 + top-2 ----------------
__global__ __launch_bounds__(256) void gate_topk(const float* __restrict__ xn2f,
                                                 const float* __restrict__ wg,
                                                 float* gv, int* gi) {
    int n = blockIdx.x, tid = threadIdx.x;
    int e = tid >> 5, lane = tid & 31;
    const float* xr = xn2f + (size_t)n * D_;
    const float* wr = wg + (size_t)e * D_;
    float s = 0.f;
    for (int d = lane; d < D_; d += 32) s += xr[d] * wr[d];
    for (int o = 16; o; o >>= 1) s += __shfl_xor(s, o);
    __shared__ float lg[8];
    if (lane == 0) lg[e] = s;
    __syncthreads();
    if (tid == 0) {
        float mx = lg[0];
        for (int i = 1; i < 8; i++) mx = fmaxf(mx, lg[i]);
        float p[8], den = 0.f;
        for (int i = 0; i < 8; i++) { p[i] = expf(lg[i] - mx); den += p[i]; }
        int i0 = 0;
        for (int i = 1; i < 8; i++) if (p[i] > p[i0]) i0 = i;
        int i1 = (i0 == 0) ? 1 : 0;
        for (int i = 0; i < 8; i++) if (i != i0 && p[i] > p[i1]) i1 = i;
        gv[n * 2]     = p[i0] / den;
        gv[n * 2 + 1] = p[i1] / den;
        gi[n * 2]     = i0;
        gi[n * 2 + 1] = i1;
    }
}

// ---------------- capacity scan: one wave per expert, token-order cumsum ----------------
__global__ __launch_bounds__(256) void route_scan(const int* __restrict__ gi, int* slot) {
    int e = threadIdx.x >> 5, lane = threadIdx.x & 31;
    int offset = 0;
    for (int base = 0; base < NT; base += 32) {
        int tok = base + lane;
        int i0 = gi[tok * 2], i1 = gi[tok * 2 + 1];
        bool m0 = (i0 == e), m1 = (i1 == e);
        bool m = m0 || m1;
        unsigned int bal = (unsigned int)__ballot(m);
        int pre = __popc(bal & ((1u << lane) - 1u));
        if (m) {
            int pos = offset + pre;
            int which = m0 ? 0 : 1;
            slot[tok * 2 + which] = (pos < CAP_) ? pos : -1;
        }
        offset += __popc(bal);
    }
}

// ---------------- scatter kept tokens into per-expert buffers ----------------
__global__ __launch_bounds__(256) void scatter_k(const int* __restrict__ gi,
                                                 const int* __restrict__ slot,
                                                 const u16* __restrict__ xn2b,
                                                 u16* __restrict__ buf) {
    int n = blockIdx.x, tid = threadIdx.x;
    const uint2* src = (const uint2*)(xn2b + (size_t)n * D_);
    for (int k = 0; k < 2; k++) {
        int s = slot[n * 2 + k];
        if (s >= 0) {
            int e = gi[n * 2 + k];
            uint2* dst = (uint2*)(buf + (size_t)(e * CAP_ + s) * D_);
            dst[tid] = src[tid];   // 256 threads * 4 bf16 = 1024
        }
    }
}

// ---------------- combine: out = h + sum_k gate_k * expert_out ----------------
__global__ __launch_bounds__(256) void combine_k(const float* __restrict__ hf,
                                                 const float* __restrict__ moef,
                                                 const float* __restrict__ gv,
                                                 const int* __restrict__ gi,
                                                 const int* __restrict__ slot,
                                                 float* __restrict__ out) {
    int n = blockIdx.x, tid = threadIdx.x;
    for (int d = tid; d < D_; d += 256) {
        float y = 0.f;
        for (int k = 0; k < 2; k++) {
            int s = slot[n * 2 + k];
            if (s >= 0) {
                int e = gi[n * 2 + k];
                y += gv[n * 2 + k] * moef[(size_t)(e * CAP_ + s) * D_ + d];
            }
        }
        out[(size_t)n * D_ + d] = hf[(size_t)n * D_ + d] + y;
    }
}

extern "C" void kernel_launch(void* const* d_in, const int* in_sizes, int n_in,
                              void* d_out, int out_size, void* d_ws, size_t ws_size,
                              hipStream_t stream) {
    const float* x      = (const float*)d_in[0];
    const float* ln1_g  = (const float*)d_in[1];
    const float* ln1_b  = (const float*)d_in[2];
    const float* w_qkv  = (const float*)d_in[3];
    const float* w_proj = (const float*)d_in[4];
    const float* ln2_g  = (const float*)d_in[5];
    const float* ln2_b  = (const float*)d_in[6];
    const float* w_gate = (const float*)d_in[7];
    const float* w1     = (const float*)d_in[8];
    const float* b1     = (const float*)d_in[9];
    const float* w2     = (const float*)d_in[10];
    const float* b2     = (const float*)d_in[11];
    float* out = (float*)d_out;

    char* ws = (char*)d_ws;
    size_t off = 0;
    auto alloc = [&](size_t bytes) -> void* {
        void* p = ws + off;
        off = (off + bytes + 255) & ~(size_t)255;
        return p;
    };
    u16*   xn1    = (u16*)  alloc((size_t)NT * D_ * 2);
    float* qkvf   = (float*)alloc((size_t)NT * 3 * D_ * 4);
    u16*   qb     = (u16*)  alloc((size_t)NT * D_ * 2);
    u16*   kbuf   = (u16*)  alloc((size_t)NT * D_ * 2);
    u16*   vbuf   = (u16*)  alloc((size_t)NT * D_ * 2);
    u16*   attnb  = (u16*)  alloc((size_t)NT * D_ * 2);
    float* hf     = (float*)alloc((size_t)NT * D_ * 4);
    u16*   xn2b   = (u16*)  alloc((size_t)NT * D_ * 2);
    float* xn2f   = (float*)alloc((size_t)NT * D_ * 4);
    float* gv     = (float*)alloc((size_t)NT * 2 * 4);
    int*   gi     = (int*)  alloc((size_t)NT * 2 * 4);
    int*   slot   = (int*)  alloc((size_t)NT * 2 * 4);
    u16*   bufb   = (u16*)  alloc((size_t)E_ * CAP_ * D_ * 2);
    u16*   h1b    = (u16*)  alloc((size_t)E_ * CAP_ * FF * 2);
    float* moef   = (float*)alloc((size_t)E_ * CAP_ * D_ * 4);
    u16*   wqkvb  = (u16*)  alloc((size_t)3 * D_ * D_ * 2);
    u16*   wprojb = (u16*)  alloc((size_t)D_ * D_ * 2);
    u16*   w1Tb   = (u16*)  alloc((size_t)E_ * FF * D_ * 2);  // (E, FF, D)
    u16*   w2Tb   = (u16*)  alloc((size_t)E_ * D_ * FF * 2);  // (E, D, FF)

    // weight prep
    cast_bf_kernel<<<4096, 256, 0, stream>>>(w_qkv, wqkvb, (long)3 * D_ * D_);
    cast_bf_kernel<<<2048, 256, 0, stream>>>(w_proj, wprojb, (long)D_ * D_);
    transpose_cast_kernel<<<8192, 256, 0, stream>>>(w1, w1Tb, E_, D_, FF); // src (E,D,FF)->(E,FF,D)
    transpose_cast_kernel<<<8192, 256, 0, stream>>>(w2, w2Tb, E_, FF, D_); // src (E,FF,D)->(E,D,FF)

    // LN1 -> bf16
    ln_kernel<<<NT, 256, 0, stream>>>(x, ln1_g, ln1_b, xn1, nullptr, D_);

    // QKV GEMM: (4096 x 3072) = xn1 (4096x1024) * w_qkv^T
    gemm_nt<0><<<dim3(3 * D_ / 64, NT / 64, 1), 128, 0, stream>>>(
        xn1, wqkvb, qkvf, nullptr, nullptr, nullptr, 3 * D_, D_, 0, 0, 0, 0);

    // RoPE + pack
    rope_pack<<<NT * H_, 64, 0, stream>>>(qkvf, qb, kbuf, vbuf);

    // causal flash attention
    flash_attn<<<dim3(T_ / 64, B_ * H_, 1), 128, 0, stream>>>(qb, kbuf, vbuf, attnb);

    // proj + residual: h = x + attn @ w_proj^T
    gemm_nt<0><<<dim3(D_ / 64, NT / 64, 1), 128, 0, stream>>>(
        attnb, wprojb, hf, nullptr, nullptr, x, D_, D_, 0, 0, 0, 0);

    // LN2 -> bf16 (for GEMM) + fp32 (for gating)
    ln_kernel<<<NT, 256, 0, stream>>>(hf, ln2_g, ln2_b, xn2b, xn2f, D_);

    // gating, capacity scan, scatter
    gate_topk<<<NT, 256, 0, stream>>>(xn2f, w_gate, gv, gi);
    route_scan<<<1, 256, 0, stream>>>(gi, slot);
    scatter_k<<<NT, 256, 0, stream>>>(gi, slot, xn2b, bufb);

    // MoE expert GEMMs (batched over E via blockIdx.z)
    gemm_nt<1><<<dim3(FF / 64, CAP_ / 64, E_), 128, 0, stream>>>(
        bufb, w1Tb, nullptr, h1b, b1, nullptr, FF, D_,
        (long)CAP_ * D_, (long)FF * D_, (long)CAP_ * FF, (long)FF);
    gemm_nt<2><<<dim3(D_ / 64, CAP_ / 64, E_), 128, 0, stream>>>(
        h1b, w2Tb, moef, nullptr, b2, nullptr, D_, FF,
        (long)CAP_ * FF, (long)D_ * FF, (long)CAP_ * D_, (long)D_);

    // combine + residual
    combine_k<<<NT, 256, 0, stream>>>(hf, moef, gv, gi, slot, out);
}
